// lstm_11055245820536
// MI455X (gfx1250) — matrix operand order repaired and checked by reading, not compiled, not run on hardware
//
#include <hip/hip_runtime.h>
#include <hip/hip_bf16.h>
#include <math.h>

// ---------------------------------------------------------------------------
// Model dims
// ---------------------------------------------------------------------------
#define BB     4
#define TTT    1024
#define EMBD   512
#define VOCAB  32000
#define NROWS  (BB * TTT)          // 4096  (head rows)
#define GROWS  (BB * (TTT - 1))    // 4092  (gate rows)

typedef __attribute__((ext_vector_type(16))) __bf16 v16bf;
typedef __attribute__((ext_vector_type(8)))  float  v8f;

__device__ __forceinline__ unsigned short f2bf(float f) {
  union { float f; unsigned u; } x; x.f = f;
  unsigned r = x.u + 0x7FFFu + ((x.u >> 16) & 1u);   // round-to-nearest-even
  return (unsigned short)(r >> 16);
}

__device__ __forceinline__ void wait_async0() {
#if __has_builtin(__builtin_amdgcn_s_wait_asynccnt)
  __builtin_amdgcn_s_wait_asynccnt(0);
#else
  asm volatile("s_wait_asynccnt 0x0" ::: "memory");
#endif
}

// ---------------------------------------------------------------------------
// Embedding gather: emb (bf16, for head), cat = [emb[:,1:], x_s] (bf16, for
// gate MLPs), x_s (f32, for beg linears)
// ---------------------------------------------------------------------------
__global__ __launch_bounds__(256) void gather_kernel(
    const int* __restrict__ x, const float* __restrict__ tok_emb,
    unsigned short* __restrict__ embb, unsigned short* __restrict__ cat,
    float* __restrict__ xs)
{
  int row = blockIdx.x;            // 0..4095
  int b = row >> 10;
  int t = row & 1023;
  int tok  = x[row];
  int tok0 = x[b << 10];
  const float* src  = tok_emb + (size_t)tok  * EMBD;
  const float* src0 = tok_emb + (size_t)tok0 * EMBD;
  for (int i = threadIdx.x; i < EMBD; i += 256) {
    float e = src[i];
    embb[(size_t)row * EMBD + i] = f2bf(e);
    if (t == 0) {
      xs[b * EMBD + i] = e;
    } else {
      size_t crow = (size_t)b * (TTT - 1) + (t - 1);
      cat[crow * (2 * EMBD) + i]        = f2bf(e);
      cat[crow * (2 * EMBD) + EMBD + i] = f2bf(src0[i]);
    }
  }
}

// ---------------------------------------------------------------------------
// WMMA GEMM: C[M,N] = act(A_bf16[M,K] @ W_f32[K,N] + bias[N])
// Tile 128x128, K-tile 64, 256 threads = 8 waves, each wave 32x64 sub-tile.
// A tile: async DMA global->LDS (ASYNCcnt), double-buffered.
// W tile: f32 streamed to VGPRs, converted to bf16, packed ds_store_b128,
//         stored transposed Bs[n][k]; also double-buffered.
// ---------------------------------------------------------------------------
#define BM 128
#define BN 128
#define BK 64
#define LSTR 72   // bf16 elems per LDS row (144B: 16B aligned, bank-skewed)

union Frag { uint4 q[2]; v16bf v; };

template<int ACT, bool OBF>   // ACT: 0=none 1=relu 2=sigmoid 3=tanh
__global__ __launch_bounds__(256) void gemm_kernel(
    const unsigned short* __restrict__ A,
    const float* __restrict__ W,
    const float* __restrict__ bias,
    void* __restrict__ Cout,
    int M, int K, int N)
{
  __shared__ __align__(16) unsigned short As[2][BM * LSTR];
  __shared__ __align__(16) unsigned short Bs[2][BN * LSTR];  // transposed: Bs[n][k]

  const int tid  = threadIdx.x;
  const int lane = tid & 31;
  const int wave = tid >> 5;   // 0..7
  const int wm   = wave & 3;   // 4 row groups of 32
  const int wn   = wave >> 2;  // 2 col groups of 64
  const int lm   = lane & 15;
  const int lg   = lane >> 4;  // 0/1
  const int kb   = lg * 8;     // A-frag K octet base

  const int bn = blockIdx.x * BN;
  const int bm = blockIdx.y * BM;

  const int nW = tid & 127;         // W staging: this thread's column
  const int kh = (tid >> 7) * 32;   // W staging: k-half (0 or 32)

  v8f acc[2][4];
  v8f zero = {0.f, 0.f, 0.f, 0.f, 0.f, 0.f, 0.f, 0.f};
#pragma unroll
  for (int mt = 0; mt < 2; ++mt)
#pragma unroll
    for (int nt = 0; nt < 4; ++nt) acc[mt][nt] = zero;

  float wpre[32];    // W tile staging (32 f32 per thread, one column)

  // Per-thread A staging chunks: ch = tid + c*256 -> row r = ch>>3, k-chunk kc
  // = (ch&7)*8 bf16 elems (16B). Rows >= M are clamped to M-1: they only feed
  // C rows >= M which are never stored.
#define ISSUE_A_ASYNC(K0, BUF)                                               \
  {                                                                          \
    _Pragma("unroll")                                                        \
    for (int c = 0; c < 4; ++c) {                                            \
      int ch = tid + c * 256;                                                \
      int r = ch >> 3, kc = (ch & 7) * 8;                                    \
      int gm = bm + r; if (gm >= M) gm = M - 1;                              \
      unsigned long long ga =                                                \
          (unsigned long long)(A + (size_t)gm * K + (K0) + kc);              \
      unsigned lo = (unsigned)(size_t)(&As[BUF][r * LSTR + kc]);             \
      asm volatile("global_load_async_to_lds_b128 %0, %1, off"               \
                   :: "v"(lo), "v"(ga) : "memory");                          \
    }                                                                        \
  }

  // --- stage tile 0
  ISSUE_A_ASYNC(0, 0)
#pragma unroll
  for (int j = 0; j < 32; ++j)
    wpre[j] = W[(size_t)(kh + j) * N + bn + nW];

  int ibuf = 0;
  for (int k0 = 0; k0 < K; k0 += BK, ibuf ^= 1) {
    wait_async0();               // my wave's DMA into As[ibuf] complete
    // --- commit staged W tile to LDS (waits loadcnt on wpre automatically)
    {
      unsigned pk[16];
#pragma unroll
      for (int j = 0; j < 16; ++j)
        pk[j] = (unsigned)f2bf(wpre[2 * j]) | ((unsigned)f2bf(wpre[2 * j + 1]) << 16);
#pragma unroll
      for (int q = 0; q < 4; ++q)
        *(uint4*)(&Bs[ibuf][nW * LSTR + kh + q * 8]) =
            make_uint4(pk[4 * q + 0], pk[4 * q + 1], pk[4 * q + 2], pk[4 * q + 3]);
    }
    __syncthreads();             // all waves' A DMA + W stores visible
    // --- stage next tile into the other buffer (overlaps WMMAs below)
    int kn = k0 + BK;
    if (kn < K) {
      ISSUE_A_ASYNC(kn, ibuf ^ 1)
#pragma unroll
      for (int j = 0; j < 32; ++j)
        wpre[j] = W[(size_t)(kn + kh + j) * N + bn + nW];
      if (kn + BK < K)
        __builtin_prefetch(W + (size_t)(kn + BK + kh) * N + bn + nW, 0, 1);
    }
    // --- compute: two 32-wide K sub-steps, 16 WMMAs per wave
#pragma unroll
    for (int ks = 0; ks < 2; ++ks) {
      Frag af[2];
#pragma unroll
      for (int mt = 0; mt < 2; ++mt) {
        // A 16x32 bf16 layout: lane holds M=lane%16, K in {kb..kb+7, kb+16..kb+23}
        const unsigned short* p = &As[ibuf][(wm * 32 + mt * 16 + lm) * LSTR + ks * 32 + kb];
        af[mt].q[0] = *(const uint4*)p;
        af[mt].q[1] = *(const uint4*)(p + 16);   // +32B -> K+16 octet
      }
#pragma unroll
      for (int nt = 0; nt < 4; ++nt) {
        // B 32x16 bf16 layout: lane holds N=lane%16, K in {kb..kb+7, kb+16..kb+23}
        Frag bfr;
        const unsigned short* p = &Bs[ibuf][(wn * 64 + nt * 16 + lm) * LSTR + ks * 32 + kb];
        bfr.q[0] = *(const uint4*)p;
        bfr.q[1] = *(const uint4*)(p + 16);
#pragma unroll
        for (int mt = 0; mt < 2; ++mt) {
          acc[mt][nt] = __builtin_amdgcn_wmma_f32_16x16x32_bf16(
              false, af[mt].v, false, bfr.v, (short)0, acc[mt][nt], false, false);
        }
      }
    }
  }
#undef ISSUE_A_ASYNC
  // --- epilogue: C/D layout m = lg*8 + v, n = lane%16
#pragma unroll
  for (int nt = 0; nt < 4; ++nt) {
    int gn = bn + wn * 64 + nt * 16 + lm;
    float bv = bias[gn];
#pragma unroll
    for (int mt = 0; mt < 2; ++mt) {
#pragma unroll
      for (int v = 0; v < 8; ++v) {
        int gm = bm + wm * 32 + mt * 16 + lg * 8 + v;
        if (gm < M) {
          float xv = acc[mt][nt][v] + bv;
          if (ACT == 1)      xv = fmaxf(xv, 0.0f);
          else if (ACT == 2) xv = 1.0f / (1.0f + expf(-xv));
          else if (ACT == 3) xv = tanhf(xv);
          if (OBF) ((unsigned short*)Cout)[(size_t)gm * N + gn] = f2bf(xv);
          else     ((float*)Cout)[(size_t)gm * N + gn] = xv;
        }
      }
    }
  }
}

// ---------------------------------------------------------------------------
// beg linears + LayerNorms (M=4, trivial): lt0 = LN(xs@Wlt+b), res0 = LN(LN(xs@Wst+b))
// ---------------------------------------------------------------------------
__device__ __forceinline__ void ln512_block(float* v, float* red,
                                            const float* g, const float* bta, int tid) {
  float a0 = v[tid], a1 = v[tid + 256];
  red[tid] = a0 + a1;
  __syncthreads();
  for (int o = 128; o > 0; o >>= 1) { if (tid < o) red[tid] += red[tid + o]; __syncthreads(); }
  float mu = red[0] * (1.0f / 512.0f);
  __syncthreads();
  float d0 = a0 - mu, d1 = a1 - mu;
  red[tid] = d0 * d0 + d1 * d1;
  __syncthreads();
  for (int o = 128; o > 0; o >>= 1) { if (tid < o) red[tid] += red[tid + o]; __syncthreads(); }
  float inv = rsqrtf(red[0] * (1.0f / 512.0f) + 1e-5f);
  __syncthreads();
  v[tid]       = d0 * inv * g[tid]       + bta[tid];
  v[tid + 256] = d1 * inv * g[tid + 256] + bta[tid + 256];
  __syncthreads();
}

__global__ __launch_bounds__(256) void beg_kernel(
    const float* __restrict__ xs,
    const float* __restrict__ wlt, const float* __restrict__ blt,
    const float* __restrict__ wst, const float* __restrict__ bst,
    const float* __restrict__ lng, const float* __restrict__ lnb,
    float* __restrict__ lt0, float* __restrict__ result)
{
  __shared__ float xrow[512];
  __shared__ float ov[512];
  __shared__ float red[256];
  int b = blockIdx.x, tid = threadIdx.x;
  for (int i = tid; i < 512; i += 256) xrow[i] = xs[b * 512 + i];
  __syncthreads();
  // lt path
  for (int h = 0; h < 2; ++h) {
    int n = tid + h * 256;
    float acc = blt[n];
    for (int k = 0; k < 512; ++k) acc = fmaf(xrow[k], wlt[(size_t)k * 512 + n], acc);
    ov[n] = acc;
  }
  __syncthreads();
  ln512_block(ov, red, lng, lnb, tid);
  lt0[b * 512 + tid]       = ov[tid];
  lt0[b * 512 + tid + 256] = ov[tid + 256];
  __syncthreads();
  // st path (double LN)
  for (int h = 0; h < 2; ++h) {
    int n = tid + h * 256;
    float acc = bst[n];
    for (int k = 0; k < 512; ++k) acc = fmaf(xrow[k], wst[(size_t)k * 512 + n], acc);
    ov[n] = acc;
  }
  __syncthreads();
  ln512_block(ov, red, lng, lnb, tid);
  ln512_block(ov, red, lng, lnb, tid);
  result[((size_t)b * TTT) * EMBD + tid]       = ov[tid];
  result[((size_t)b * TTT) * EMBD + tid + 256] = ov[tid + 256];
}

// ---------------------------------------------------------------------------
// Sequential scan: 4 blocks (one per batch), 512 threads (one per feature).
// Per-step LNs via wave32 shfl_xor + 16-wave LDS stage (2 barriers/reduction).
// ---------------------------------------------------------------------------
__device__ __forceinline__ float blk_sum512(float v, float* red, int tid) {
#pragma unroll
  for (int o = 16; o > 0; o >>= 1) v += __shfl_xor(v, o, 32);
  if ((tid & 31) == 0) red[tid >> 5] = v;
  __syncthreads();
  float s = 0.f;
#pragma unroll
  for (int i = 0; i < 16; ++i) s += red[i];
  __syncthreads();
  return s;
}

__global__ __launch_bounds__(512) void scan_kernel(
    const float* __restrict__ gm_, const float* __restrict__ gs_,
    const float* __restrict__ gt_, const float* __restrict__ gl_,
    const float* __restrict__ lt0,
    const float* __restrict__ lng, const float* __restrict__ lnb,
    float* __restrict__ result)
{
  __shared__ float red[16];
  int b = blockIdx.x, i = threadIdx.x;
  float gain = lng[i], beta = lnb[i];
  float lt = lt0[b * 512 + i];
  for (int t = 0; t < TTT - 1; ++t) {
    size_t idx = ((size_t)b * (TTT - 1) + t) * EMBD + i;
    float m = gm_[idx], s = gs_[idx], th = gt_[idx], l = gl_[idx];
    float ltn = fmaf(m, lt, s * th);
    float st  = tanhf(ltn) * l;
    // carry = LN(ltn)
    float mu  = blk_sum512(ltn, red, i) * (1.f / 512.f);
    float d   = ltn - mu;
    float var = blk_sum512(d * d, red, i) * (1.f / 512.f);
    lt = d * rsqrtf(var + 1e-5f) * gain + beta;
    // out = LN(st)
    float mu2  = blk_sum512(st, red, i) * (1.f / 512.f);
    float d2   = st - mu2;
    float var2 = blk_sum512(d2 * d2, red, i) * (1.f / 512.f);
    result[((size_t)b * TTT + t + 1) * EMBD + i] =
        d2 * rsqrtf(var2 + 1e-5f) * gain + beta;
  }
}

// ---------------------------------------------------------------------------
// Host: input order = x, tok_emb, beg_lt(w,b), beg_st(w,b), mul_p(5x w,b),
// sig_p(3x), tanh_p(3x), ltst_p(3x), ln_g, ln_b, head_p(3x)
// ---------------------------------------------------------------------------
extern "C" void kernel_launch(void* const* d_in, const int* in_sizes, int n_in,
                              void* d_out, int out_size, void* d_ws, size_t ws_size,
                              hipStream_t stream)
{
  (void)in_sizes; (void)n_in; (void)out_size; (void)ws_size;

  const int*   x       = (const int*)d_in[0];
  const float* tok_emb = (const float*)d_in[1];
  const float* lng     = (const float*)d_in[34];
  const float* lnb     = (const float*)d_in[35];
  auto W = [&](int i) { return (const float*)d_in[i]; };

  char* wsp = (char*)d_ws;
  size_t off = 0;
  auto alloc = [&](size_t bytes) -> void* {
    void* p = wsp + off;
    off += (bytes + 255) & ~(size_t)255;
    return p;
  };
  unsigned short* cat   = (unsigned short*)alloc((size_t)GROWS * 1024 * 2);
  unsigned short* embb  = (unsigned short*)alloc((size_t)NROWS * 512 * 2);
  unsigned short* buf1  = (unsigned short*)alloc((size_t)4096 * 4096 * 2);
  unsigned short* buf2  = (unsigned short*)alloc((size_t)4096 * 4096 * 2);
  float* gmul  = (float*)alloc((size_t)GROWS * 512 * 4);
  float* gsig  = (float*)alloc((size_t)GROWS * 512 * 4);
  float* gtanh = (float*)alloc((size_t)GROWS * 512 * 4);
  float* glt   = (float*)alloc((size_t)GROWS * 512 * 4);
  float* xs    = (float*)alloc(4 * 512 * 4);
  float* lt0   = (float*)alloc(4 * 512 * 4);

  float* logits = (float*)d_out;
  float* result = (float*)d_out + (size_t)NROWS * VOCAB;

  gather_kernel<<<dim3(NROWS), dim3(256), 0, stream>>>(x, tok_emb, embb, cat, xs);
  beg_kernel<<<dim3(BB), dim3(256), 0, stream>>>(
      xs, W(2), W(3), W(4), W(5), lng, lnb, lt0, result);

  auto G = [&](int act, bool obf, const unsigned short* Aa, int wi, int bi,
               void* Cc, int M, int K, int N) {
    dim3 grid(N / BN, (M + BM - 1) / BM), blk(256);
    if (obf)             gemm_kernel<1, true ><<<grid, blk, 0, stream>>>(Aa, W(wi), W(bi), Cc, M, K, N);
    else if (act == 0)   gemm_kernel<0, false><<<grid, blk, 0, stream>>>(Aa, W(wi), W(bi), Cc, M, K, N);
    else if (act == 2)   gemm_kernel<2, false><<<grid, blk, 0, stream>>>(Aa, W(wi), W(bi), Cc, M, K, N);
    else                 gemm_kernel<3, false><<<grid, blk, 0, stream>>>(Aa, W(wi), W(bi), Cc, M, K, N);
  };

  // mul gate: 1024->2048->4096->2048->1024->512 (sigmoid)
  G(1, true,  cat,   6,  7, buf1, GROWS, 1024, 2048);
  G(1, true,  buf1,  8,  9, buf2, GROWS, 2048, 4096);
  G(1, true,  buf2, 10, 11, buf1, GROWS, 4096, 2048);
  G(1, true,  buf1, 12, 13, buf2, GROWS, 2048, 1024);
  G(2, false, buf2, 14, 15, gmul, GROWS, 1024, 512);
  // sig gate: 1024->2048->1024->512 (sigmoid)
  G(1, true,  cat,  16, 17, buf1, GROWS, 1024, 2048);
  G(1, true,  buf1, 18, 19, buf2, GROWS, 2048, 1024);
  G(2, false, buf2, 20, 21, gsig, GROWS, 1024, 512);
  // tanh gate
  G(1, true,  cat,  22, 23, buf1, GROWS, 1024, 2048);
  G(1, true,  buf1, 24, 25, buf2, GROWS, 2048, 1024);
  G(3, false, buf2, 26, 27, gtanh, GROWS, 1024, 512);
  // ltst gate (sigmoid)
  G(1, true,  cat,  28, 29, buf1, GROWS, 1024, 2048);
  G(1, true,  buf1, 30, 31, buf2, GROWS, 2048, 1024);
  G(2, false, buf2, 32, 33, glt, GROWS, 1024, 512);

  scan_kernel<<<dim3(BB), dim3(512), 0, stream>>>(
      gmul, gsig, gtanh, glt, lt0, lng, lnb, result);

  // head: 512->1024->2048->32000
  G(1, true,  embb, 36, 37, buf1, NROWS, 512, 1024);
  G(1, true,  buf1, 38, 39, buf2, NROWS, 1024, 2048);
  G(0, false, buf2, 40, 41, logits, NROWS, 2048, VOCAB);
}
